// AttentionHead_87703232184451
// MI455X (gfx1250) — compile-verified, matrix operand order checked by reading, not run on hardware
//
#include <hip/hip_runtime.h>
#include <hip/hip_bf16.h>

// ---------------------------------------------------------------------------
// res[b,s,d] = scale * (q @ (kT @ v))[b,s,d] - 1e9 * sum_{t>s} v[b,t,d]
//   q,k,v = x @ W^T  (x:[B,S,H] f32, W:[D,H] f32), B=4 S=4096 H=1024 D=128
// All GEMMs via V_WMMA_F32_16X16X32_BF16 (f32 accumulate).
// ---------------------------------------------------------------------------

typedef __attribute__((ext_vector_type(16))) __bf16 v16bf;
typedef __attribute__((ext_vector_type(8)))  __bf16 v8bf;
typedef __attribute__((ext_vector_type(8)))  float  v8f;
typedef __attribute__((ext_vector_type(4)))  float  v4f;

#define WMMA_BF16(a, b, c) \
  __builtin_amdgcn_wmma_f32_16x16x32_bf16(false, (a), false, (b), (short)0, (c), false, false)

#define NB  4
#define NS  4096
#define NH  1024
#define ND  128
#define NQKV 384   // q|k|v concatenated along the output-feature axis

static __device__ __forceinline__ v8f zero8() {
  v8f z;
#pragma unroll
  for (int i = 0; i < 8; i++) z[i] = 0.0f;
  return z;
}

// A/B fragment from a row-contiguous bf16 matrix: lane holds two 8-elem runs,
// K = hi*8 + {0..7} and K = 16 + hi*8 + {0..7}   (ISA 7.12.2, 16-bit 16x32 A)
static __device__ __forceinline__ v16bf frag_rows_bf16(const __bf16* p, int hi) {
  v8bf r0 = *(const v8bf*)(p + hi * 8);
  v8bf r1 = *(const v8bf*)(p + 16 + hi * 8);
  v16bf f;
#pragma unroll
  for (int i = 0; i < 8; i++) { f[i] = r0[i]; f[8 + i] = r1[i]; }
  return f;
}

// Same fragment but source is f32 (x matrix): convert in-register to bf16.
static __device__ __forceinline__ v16bf frag_rows_f32(const float* p, int hi) {
  v4f a0 = *(const v4f*)(p + hi * 8);
  v4f a1 = *(const v4f*)(p + hi * 8 + 4);
  v4f b0 = *(const v4f*)(p + 16 + hi * 8);
  v4f b1 = *(const v4f*)(p + 16 + hi * 8 + 4);
  v16bf f;
#pragma unroll
  for (int i = 0; i < 4; i++) {
    f[i]      = (__bf16)a0[i];
    f[4 + i]  = (__bf16)a1[i];
    f[8 + i]  = (__bf16)b0[i];
    f[12 + i] = (__bf16)b1[i];
  }
  return f;
}

// B fragment gathered down a column of a row-major matrix (stride = row pitch).
// Element e -> K = kBase + (e&8)*2 + hi*8 + (e&7). Lane-coalesced (16 lanes x 2B).
static __device__ __forceinline__ v16bf frag_col_strided(const __bf16* col, int stride,
                                                         int kBase, int hi) {
  v16bf f;
#pragma unroll
  for (int e = 0; e < 16; e++) {
    int K = kBase + ((e & 8) << 1) + hi * 8 + (e & 7);
    f[e] = col[(size_t)K * stride];
  }
  return f;
}

// ---------------------------------------------------------------------------
// Kernel 1: Wq|Wk|Wv (f32 [128,1024] each) -> Wb bf16 [384,1024]
// ---------------------------------------------------------------------------
__global__ void k_convert_w(const float* __restrict__ Wq, const float* __restrict__ Wk,
                            const float* __restrict__ Wv, __bf16* __restrict__ Wb) {
  int n = blockIdx.x * blockDim.x + threadIdx.x;           // 0 .. 393215
  const float* src = (n < 131072) ? Wq : (n < 262144 ? Wk : Wv);
  Wb[n] = (__bf16)src[n & 131071];
}

// ---------------------------------------------------------------------------
// Kernel 2: QKV projection. qkv[g, c] = sum_h x[g,h] * Wb[c,h]  (bf16 out)
// grid (3 colBlocks, 128 rowBlocks), 256 thr = 8 waves; wave: 16 rows x 128 cols
// ---------------------------------------------------------------------------
__global__ void __launch_bounds__(256) k_qkv(const float* __restrict__ x,
                                             const __bf16* __restrict__ Wb,
                                             __bf16* __restrict__ qkv) {
  const int lane = threadIdx.x & 31;
  const int wv   = threadIdx.x >> 5;
  const int hi   = lane >> 4;
  const int lr   = lane & 15;
  const int colBlk  = blockIdx.x;                 // 0..2
  const int rowBase = blockIdx.y * 128 + wv * 16; // wave's 16-row strip
  const float* xrow = x + (size_t)(rowBase + lr) * NH;

  v8f acc[8];
#pragma unroll
  for (int t = 0; t < 8; t++) acc[t] = zero8();

  for (int kB = 0; kB < NH; kB += 32) {
    if (kB + 64 < NH) __builtin_prefetch(xrow + kB + 64, 0, 0);
    v16bf a = frag_rows_f32(xrow + kB, hi);       // A: x rows, f32->bf16 inline
#pragma unroll
    for (int t = 0; t < 8; t++) {
      int c = colBlk * 128 + t * 16 + lr;         // B: W row c is contiguous in h
      v16bf b = frag_rows_bf16(Wb + (size_t)c * NH + kB, hi);
      acc[t] = WMMA_BF16(a, b, acc[t]);
    }
  }
#pragma unroll
  for (int t = 0; t < 8; t++) {
    int c = colBlk * 128 + t * 16 + lr;
#pragma unroll
    for (int r = 0; r < 8; r++) {
      int row = rowBase + 8 * hi + r;             // C layout: VGPR r -> row r+hi*8
      qkv[(size_t)row * NQKV + c] = (__bf16)acc[t][r];
    }
  }
}

// ---------------------------------------------------------------------------
// Kernels 3-5: Vsuffix[b,s,d] = sum_{t>s} v[b,t,d]   (chunked suffix scan)
// v[b,t,d] lives at qkv[(b*S+t)*384 + 256 + d]
// ---------------------------------------------------------------------------
__global__ void k_vchunk(const __bf16* __restrict__ qkv, float* __restrict__ cs) {
  int cx = blockIdx.x, b = blockIdx.y, d = threadIdx.x;    // 16 x 4 blocks, 128 thr
  const __bf16* vb = qkv + (size_t)b * NS * NQKV + 256 + d;
  float acc = 0.f;
  int t0 = cx * 256;
  for (int t = 0; t < 256; t++) acc += (float)vb[(size_t)(t0 + t) * NQKV];
  cs[(b * 16 + cx) * ND + d] = acc;
}

__global__ void k_vscan(float* __restrict__ cs) {
  int b = blockIdx.x, d = threadIdx.x;                     // turn chunk sums into
  float run = 0.f;                                         // exclusive suffix sums
  for (int c = 15; c >= 0; c--) {
    int idx = (b * 16 + c) * ND + d;
    float t = cs[idx];
    cs[idx] = run;
    run += t;
  }
}

__global__ void k_vfill(const __bf16* __restrict__ qkv, const float* __restrict__ cs,
                        float* __restrict__ vsuf) {
  int cx = blockIdx.x, b = blockIdx.y, d = threadIdx.x;
  const __bf16* vb = qkv + (size_t)b * NS * NQKV + 256 + d;
  float run = cs[(b * 16 + cx) * ND + d];                  // suffix beyond this chunk
  for (int t = cx * 256 + 255; t >= cx * 256; t--) {
    vsuf[((size_t)b * NS + t) * ND + d] = run;
    run += (float)vb[(size_t)t * NQKV];
  }
}

// ---------------------------------------------------------------------------
// Kernel 6: Msc[b] = scale * (kT[b] @ v[b])  -> bf16 [128,128] per batch
// kT row i = flat(k[b]) elements [i*4096, i*4096+4096); grid 4, 8 waves
// ---------------------------------------------------------------------------
__global__ void __launch_bounds__(256) k_m(const __bf16* __restrict__ qkv,
                                           __bf16* __restrict__ Msc) {
  const int lane = threadIdx.x & 31;
  const int wv   = threadIdx.x >> 5;
  const int hi   = lane >> 4;
  const int lr   = lane & 15;
  const int b    = blockIdx.x;
  const __bf16* kfl = qkv + (size_t)b * NS * NQKV + 128;   // k block, pitch 384
  const __bf16* vc0 = qkv + (size_t)b * NS * NQKV + 256;   // v block, pitch 384
  const int i = wv * 16 + lr;                              // A-matrix row (kT row)

  v8f acc[8];
#pragma unroll
  for (int t = 0; t < 8; t++) acc[t] = zero8();

  for (int kB = 0; kB < NS; kB += 32) {
    // A fragment: two 8-elem runs of kT row i (each run stays inside one k row)
    int n0 = i * NS + kB + hi * 8;
    int n1 = n0 + 16;
    v8bf r0 = *(const v8bf*)(kfl + (size_t)(n0 >> 7) * NQKV + (n0 & 127));
    v8bf r1 = *(const v8bf*)(kfl + (size_t)(n1 >> 7) * NQKV + (n1 & 127));
    v16bf a;
#pragma unroll
    for (int e = 0; e < 8; e++) { a[e] = r0[e]; a[8 + e] = r1[e]; }
#pragma unroll
    for (int t = 0; t < 8; t++) {
      v16bf bf = frag_col_strided(vc0 + t * 16 + lr, NQKV, kB, hi);
      acc[t] = WMMA_BF16(a, bf, acc[t]);
    }
  }
  const float scale = 0.08838834764831845f;                // 1/sqrt(128), folded in
#pragma unroll
  for (int t = 0; t < 8; t++) {
#pragma unroll
    for (int r = 0; r < 8; r++) {
      int row = wv * 16 + 8 * hi + r;
      Msc[(size_t)b * ND * ND + row * ND + (t * 16 + lr)] = (__bf16)(scale * acc[t][r]);
    }
  }
}

// ---------------------------------------------------------------------------
// Kernel 7: out[g,d] = (q[g,:] @ Msc[b]) - 1e9 * vsuf[g,d]   (f32 out)
// grid 128 blocks x 8 waves, wave: 16 rows x 128 cols, K = 128 (4 WMMA steps)
// ---------------------------------------------------------------------------
__global__ void __launch_bounds__(256) k_out(const __bf16* __restrict__ qkv,
                                             const __bf16* __restrict__ Msc,
                                             const float* __restrict__ vsuf,
                                             float* __restrict__ out) {
  const int lane = threadIdx.x & 31;
  const int wv   = threadIdx.x >> 5;
  const int hi   = lane >> 4;
  const int lr   = lane & 15;
  const int rowBase = blockIdx.x * 128 + wv * 16;
  const int b = rowBase >> 12;                             // batch (4096 rows each)
  const __bf16* qrow = qkv + (size_t)(rowBase + lr) * NQKV; // q at col offset 0
  const __bf16* Mb   = Msc + (size_t)b * ND * ND;

  v8f acc[8];
#pragma unroll
  for (int t = 0; t < 8; t++) acc[t] = zero8();

#pragma unroll
  for (int kB = 0; kB < ND; kB += 32) {
    v16bf a = frag_rows_bf16(qrow + kB, hi);
#pragma unroll
    for (int t = 0; t < 8; t++) {
      v16bf bf = frag_col_strided(Mb + t * 16 + lr, ND, kB, hi);
      acc[t] = WMMA_BF16(a, bf, acc[t]);
    }
  }
#pragma unroll
  for (int t = 0; t < 8; t++) {
    int col = t * 16 + lr;
#pragma unroll
    for (int r = 0; r < 8; r++) {
      size_t g = (size_t)(rowBase + 8 * hi + r) * ND + col;
      out[g] = acc[t][r] - 1e9f * vsuf[g];                 // fused causal-mask term
    }
  }
}

// ---------------------------------------------------------------------------
extern "C" void kernel_launch(void* const* d_in, const int* in_sizes, int n_in,
                              void* d_out, int out_size, void* d_ws, size_t ws_size,
                              hipStream_t stream) {
  (void)in_sizes; (void)n_in; (void)out_size; (void)ws_size;
  const float* x  = (const float*)d_in[0];
  const float* Wq = (const float*)d_in[1];
  const float* Wk = (const float*)d_in[2];
  const float* Wv = (const float*)d_in[3];
  float* out = (float*)d_out;

  char* ws = (char*)d_ws;                                  // < 24 MB total
  __bf16* Wb   = (__bf16*)(ws);                            // 0.75 MB  [384,1024]
  __bf16* qkv  = (__bf16*)(ws + (size_t)(1u  << 20));      // 12 MB    [16384,384]
  float*  vsuf = (float*) (ws + (size_t)(14u << 20));      // 8 MB     [16384,128]
  float*  cs   = (float*) (ws + (size_t)(23u << 20));      // 32 KB    chunk sums
  __bf16* Msc  = (__bf16*)(ws + (size_t)(23u << 20) + 65536); // 128 KB [4,128,128]

  k_convert_w<<<dim3(1536), dim3(256), 0, stream>>>(Wq, Wk, Wv, Wb);
  k_qkv      <<<dim3(3, 128), dim3(256), 0, stream>>>(x, Wb, qkv);
  k_vchunk   <<<dim3(16, 4), dim3(128), 0, stream>>>(qkv, cs);
  k_vscan    <<<dim3(4),     dim3(128), 0, stream>>>(cs);
  k_vfill    <<<dim3(16, 4), dim3(128), 0, stream>>>(qkv, cs, vsuf);
  k_m        <<<dim3(4),     dim3(256), 0, stream>>>(qkv, Msc);
  k_out      <<<dim3(128),   dim3(256), 0, stream>>>(qkv, Msc, vsuf, out);
}
